// EncodeProcessDecode_82274393522653
// MI455X (gfx1250) — compile-verified, matrix operand order checked by reading, not run on hardware
//
#include <hip/hip_runtime.h>
#include <hip/hip_bf16.h>
#include <stdint.h>
#include <math.h>

#define NN_ 10000   // nodes
#define NE_ 60000   // edges
#define LL  128     // latent
#define SS  8       // message passing steps
#define TWD 5
#define OUTF 3

typedef __attribute__((ext_vector_type(16))) __bf16 v16bf;
typedef __attribute__((ext_vector_type(8)))  float v8f;

// ---------- helpers ----------

__device__ __forceinline__ uint16_t f2bf(float f) {
    union { float f; uint32_t u; } x; x.f = f;
    uint32_t u = x.u;
    if ((u & 0x7F800000u) == 0x7F800000u && (u & 0x007FFFFFu)) // NaN
        return (uint16_t)((u >> 16) | 0x40);
    return (uint16_t)((u + 0x7FFFu + ((u >> 16) & 1u)) >> 16);
}

// K index held by register-slot j (0..15) for lane-half hi (0 = lanes0-15, 1 = lanes16-31)
// per CDNA5 ISA 16-bit A/B fragment layout (16x32 MxK):
//   lanes 0-15 : K = 0..7 (v0-v3), 16..23 (v4-v7)
//   lanes 16-31: K = 8..15,        24..31
__device__ __forceinline__ int kmap(int j, int hi) {
    return j + (j >= 8 ? 8 : 0) + (hi ? 8 : 0);
}

__device__ __forceinline__ v8f wmma_bf16(const uint4* aq, const uint4* bq, v8f c) {
    union { v16bf v; uint4 q[2]; } a, b;
    a.q[0] = aq[0]; a.q[1] = aq[1];
    b.q[0] = bq[0]; b.q[1] = bq[1];
    return __builtin_amdgcn_wmma_f32_16x16x32_bf16(false, a.v, false, b.v,
                                                   (short)0, c, false, false);
}

// ---------- weight packing: fp32 [K,128] -> bf16 WMMA-B fragment layout ----------
// dst[((kc*128 + n)*2 + h)*16 + j] = bf16( src[(kc*32 + kmap(j,h))*128 + n] )
__global__ void pack_w_kernel(const float* __restrict__ src, uint16_t* __restrict__ dst, int K) {
    int idx = blockIdx.x * blockDim.x + threadIdx.x;
    if (idx >= K * LL) return;
    int j  =  idx        & 15;
    int h  = (idx >> 4)  & 1;
    int n  = (idx >> 5)  & 127;
    int kc =  idx >> 12;
    int k  = kc * 32 + kmap(j, h);
    dst[idx] = f2bf(src[(size_t)k * LL + n]);
}

// ---------- encoder layer-1 (K=7, VALU) ----------

__global__ void enc_node_l1_kernel(const float* __restrict__ disp, const float* __restrict__ chem,
                                   const int* __restrict__ ntype,
                                   const float* __restrict__ mean, const float* __restrict__ std_,
                                   const float* __restrict__ W1, const float* __restrict__ b1,
                                   uint16_t* __restrict__ H) {
    __shared__ float f[8];
    int n = blockIdx.x, t = threadIdx.x;
    if (t == 0) {
        float raw[7];
        raw[0] = disp[n * 2 + 0];
        raw[1] = disp[n * 2 + 1];
        raw[2] = chem[n];
        int ty = ntype[n];
        #pragma unroll
        for (int i = 0; i < 4; i++) raw[3 + i] = (ty == i) ? 1.0f : 0.0f;
        #pragma unroll
        for (int i = 0; i < 7; i++) f[i] = (raw[i] - mean[i]) / std_[i];
    }
    __syncthreads();
    float acc = b1[t];
    #pragma unroll
    for (int k = 0; k < 7; k++) acc += f[k] * W1[k * LL + t];
    H[(size_t)n * LL + t] = f2bf(acc > 0.f ? acc : 0.f);
}

__global__ void enc_edge_l1_kernel(const float* __restrict__ mpos, const float* __restrict__ disp,
                                   const float* __restrict__ chem,
                                   const int* __restrict__ send, const int* __restrict__ recv,
                                   const float* __restrict__ mean, const float* __restrict__ std_,
                                   const float* __restrict__ W1, const float* __restrict__ b1,
                                   uint16_t* __restrict__ H) {
    __shared__ float f[8];
    int e = blockIdx.x, t = threadIdx.x;
    if (t == 0) {
        int s = send[e], r = recv[e];
        float rx = mpos[s * 2 + 0] - mpos[r * 2 + 0];
        float ry = mpos[s * 2 + 1] - mpos[r * 2 + 1];
        float nrm = sqrtf(rx * rx + ry * ry);
        float dgx = disp[s * 2 + 0] - disp[r * 2 + 0];
        float dgy = disp[s * 2 + 1] - disp[r * 2 + 1];
        float cg  = chem[s] - chem[r];
        float raw[7] = { rx, ry, nrm, dgx, dgy, nrm, cg };
        #pragma unroll
        for (int i = 0; i < 7; i++) f[i] = (raw[i] - mean[i]) / std_[i];
    }
    __syncthreads();
    float acc = b1[t];
    #pragma unroll
    for (int k = 0; k < 7; k++) acc += f[k] * W1[k * LL + t];
    H[(size_t)e * LL + t] = f2bf(acc > 0.f ? acc : 0.f);
}

// ---------- layer-1 WMMA GEMMs with fused concat-gather (no materialized Xe/Xn) ----------
// Each block: 16 rows x 128 cols, 8 waves x one 16x16 tile, K fully unrolled.

// edge: A row = [ nlbf[send[e]] | nlbf[recv[e]] | elbf[e] ]  (K = 384)
__global__ void gemm_edge_l1_kernel(const uint16_t* __restrict__ nlbf,
                                    const uint16_t* __restrict__ elbf,
                                    const int* __restrict__ send, const int* __restrict__ recv,
                                    const uint16_t* __restrict__ Wp,
                                    const float* __restrict__ bias,
                                    uint16_t* __restrict__ Y) {
    int rowbase = blockIdx.x * 16;
    int wave = threadIdx.x >> 5, lane = threadIdx.x & 31;
    int l15 = lane & 15, hi = lane >> 4;
    int col = wave * 16 + l15;
    int row = rowbase + l15;
    int s = send[row], r = recv[row];
    const uint16_t* seg0 = nlbf + (size_t)s   * LL + hi * 8;
    const uint16_t* seg1 = nlbf + (size_t)r   * LL + hi * 8;
    const uint16_t* seg2 = elbf + (size_t)row * LL + hi * 8;
    v8f c;
    float bv = bias[col];
    #pragma unroll
    for (int v = 0; v < 8; v++) c[v] = bv;
    #pragma unroll
    for (int kc = 0; kc < 12; kc++) {
        const uint16_t* p = (kc < 4 ? seg0 : (kc < 8 ? seg1 : seg2)) + (kc & 3) * 32;
        uint4 aq[2] = { *(const uint4*)p, *(const uint4*)(p + 16) };
        const uint16_t* wp = Wp + ((size_t)(kc * LL + col) * 2 + hi) * 16;
        uint4 bq[2] = { *(const uint4*)wp, *(const uint4*)(wp + 8) };
        c = wmma_bf16(aq, bq, c);
    }
    int m0 = hi * 8;
    #pragma unroll
    for (int v = 0; v < 8; v++) {
        float y = c[v] > 0.f ? c[v] : 0.f;
        Y[(size_t)(rowbase + m0 + v) * LL + col] = f2bf(y);
    }
}

// node: A row = [ nlbf[n] | aggrbf[n] ]  (K = 256)
__global__ void gemm_node_l1_kernel(const uint16_t* __restrict__ nlbf,
                                    const uint16_t* __restrict__ aggrbf,
                                    const uint16_t* __restrict__ Wp,
                                    const float* __restrict__ bias,
                                    uint16_t* __restrict__ Y) {
    int rowbase = blockIdx.x * 16;
    int wave = threadIdx.x >> 5, lane = threadIdx.x & 31;
    int l15 = lane & 15, hi = lane >> 4;
    int col = wave * 16 + l15;
    int row = rowbase + l15;
    const uint16_t* seg0 = nlbf   + (size_t)row * LL + hi * 8;
    const uint16_t* seg1 = aggrbf + (size_t)row * LL + hi * 8;
    v8f c;
    float bv = bias[col];
    #pragma unroll
    for (int v = 0; v < 8; v++) c[v] = bv;
    #pragma unroll
    for (int kc = 0; kc < 8; kc++) {
        const uint16_t* p = (kc < 4 ? seg0 : seg1) + (kc & 3) * 32;
        uint4 aq[2] = { *(const uint4*)p, *(const uint4*)(p + 16) };
        const uint16_t* wp = Wp + ((size_t)(kc * LL + col) * 2 + hi) * 16;
        uint4 bq[2] = { *(const uint4*)wp, *(const uint4*)(wp + 8) };
        c = wmma_bf16(aq, bq, c);
    }
    int m0 = hi * 8;
    #pragma unroll
    for (int v = 0; v < 8; v++) {
        float y = c[v] > 0.f ? c[v] : 0.f;
        Y[(size_t)(rowbase + m0 + v) * LL + col] = f2bf(y);
    }
}

// ---------- layer-2 WMMA GEMM: out = resid + LN(relu(X@W+b))*g+be, + edge scatter-add ----------

__global__ void gemm_relu_ln_kernel(const uint16_t* __restrict__ X, const uint16_t* __restrict__ Wp,
                                    const float* __restrict__ bias,
                                    const float* __restrict__ g, const float* __restrict__ be,
                                    const float* __restrict__ resid,      // nullable
                                    float* __restrict__ out_f32, uint16_t* __restrict__ out_bf16,
                                    float* __restrict__ aggr,             // nullable (edge mode)
                                    const int* __restrict__ recv) {       // nullable (edge mode)
    __shared__ float tile[16 * 132];
    __shared__ float mu_s[16], rs_s[16];
    int rowbase = blockIdx.x * 16;
    int wave = threadIdx.x >> 5, lane = threadIdx.x & 31;
    int l15 = lane & 15, hi = lane >> 4;
    int col = wave * 16 + l15;
    v8f c;
    float bv = bias[col];
    #pragma unroll
    for (int v = 0; v < 8; v++) c[v] = bv;
    const uint16_t* xrow = X + (size_t)(rowbase + l15) * LL + hi * 8;
    #pragma unroll
    for (int kc = 0; kc < 4; kc++) {
        uint4 aq[2] = { *(const uint4*)(xrow + kc * 32), *(const uint4*)(xrow + kc * 32 + 16) };
        const uint16_t* wp = Wp + ((size_t)(kc * LL + col) * 2 + hi) * 16;
        uint4 bq[2] = { *(const uint4*)wp, *(const uint4*)(wp + 8) };
        c = wmma_bf16(aq, bq, c);
    }
    int m0 = hi * 8;
    #pragma unroll
    for (int v = 0; v < 8; v++) {
        float y = c[v] > 0.f ? c[v] : 0.f;
        tile[(m0 + v) * 132 + col] = y;
    }
    __syncthreads();
    int t = threadIdx.x;
    if (t < 16) {
        float s = 0.f, s2 = 0.f;
        for (int cc = 0; cc < 128; cc++) {
            float v = tile[t * 132 + cc];
            s += v; s2 += v * v;
        }
        float m = s * (1.0f / 128.0f);
        float var = s2 * (1.0f / 128.0f) - m * m;
        mu_s[t] = m;
        rs_s[t] = rsqrtf(var + 1e-5f);
    }
    __syncthreads();
    #pragma unroll
    for (int i = 0; i < 8; i++) {
        int e = t + 256 * i;
        int r = e >> 7, cc = e & 127;
        float y = (tile[r * 132 + cc] - mu_s[r]) * rs_s[r] * g[cc] + be[cc];
        int grow = rowbase + r;
        size_t oi = (size_t)grow * LL + cc;
        float o = y;
        if (resid) o += resid[oi];
        out_f32[oi] = o;
        out_bf16[oi] = f2bf(o);
        if (aggr) atomicAdd(&aggr[(size_t)recv[grow] * LL + cc], y);
    }
}

// ---------- misc ----------

__global__ void zero_f32_kernel(float* __restrict__ p, int n) {
    int i = blockIdx.x * blockDim.x + threadIdx.x;
    if (i < n) p[i] = 0.f;
}

__global__ void cvt_bf16_kernel(const float* __restrict__ src, uint16_t* __restrict__ dst, int n) {
    int i = blockIdx.x * blockDim.x + threadIdx.x;
    if (i < n) dst[i] = f2bf(src[i]);
}

// ---------- decoder ----------

__global__ void decoder_kernel(const float* __restrict__ nl,
                               const float* __restrict__ W1, const float* __restrict__ b1,
                               const float* __restrict__ W2, const float* __restrict__ b2,
                               float* __restrict__ out) {
    int n = blockIdx.x * blockDim.x + threadIdx.x;
    if (n >= NN_) return;
    float h[8];
    #pragma unroll
    for (int o = 0; o < 8; o++) {
        float acc = b1[o];
        for (int k = 0; k < LL; k++) acc += nl[(size_t)n * LL + k] * W1[k * 8 + o];
        h[o] = acc / (1.0f + expf(-acc));   // swish
    }
    #pragma unroll
    for (int j = 0; j < OUTF * TWD; j++) {
        float acc = b2[j];
        #pragma unroll
        for (int o = 0; o < 8; o++) acc += h[o] * W2[o * (OUTF * TWD) + j];
        int tw = j / OUTF, oo = j % OUTF;
        out[(size_t)tw * NN_ * OUTF + (size_t)n * OUTF + oo] = acc * (float)(tw + 1);
    }
}

// ---------- host launch ----------

extern "C" void kernel_launch(void* const* d_in, const int* in_sizes, int n_in,
                              void* d_out, int out_size, void* d_ws, size_t ws_size,
                              hipStream_t stream) {
    (void)in_sizes; (void)n_in; (void)out_size; (void)ws_size;

    const float* mesh_pos  = (const float*)d_in[0];
    const float* disp      = (const float*)d_in[1];
    const float* chem      = (const float*)d_in[2];
    const int*   ntype     = (const int*)d_in[3];
    const int*   send      = (const int*)d_in[4];
    const int*   recv      = (const int*)d_in[5];
    const float* node_mean = (const float*)d_in[6];
    const float* node_std  = (const float*)d_in[7];
    const float* edge_mean = (const float*)d_in[8];
    const float* edge_std  = (const float*)d_in[9];
    const float* en_W1 = (const float*)d_in[10]; const float* en_b1 = (const float*)d_in[11];
    const float* en_W2 = (const float*)d_in[12]; const float* en_b2 = (const float*)d_in[13];
    const float* en_g  = (const float*)d_in[14]; const float* en_be = (const float*)d_in[15];
    const float* ee_W1 = (const float*)d_in[16]; const float* ee_b1 = (const float*)d_in[17];
    const float* ee_W2 = (const float*)d_in[18]; const float* ee_b2 = (const float*)d_in[19];
    const float* ee_g  = (const float*)d_in[20]; const float* ee_be = (const float*)d_in[21];
    const float* be_W1 = (const float*)d_in[22]; const float* be_b1 = (const float*)d_in[23];
    const float* be_W2 = (const float*)d_in[24]; const float* be_b2 = (const float*)d_in[25];
    const float* be_g  = (const float*)d_in[26]; const float* be_be = (const float*)d_in[27];
    const float* bn_W1 = (const float*)d_in[28]; const float* bn_b1 = (const float*)d_in[29];
    const float* bn_W2 = (const float*)d_in[30]; const float* bn_b2 = (const float*)d_in[31];
    const float* bn_g  = (const float*)d_in[32]; const float* bn_be = (const float*)d_in[33];
    const float* dec_W1 = (const float*)d_in[34]; const float* dec_b1 = (const float*)d_in[35];
    const float* dec_W2 = (const float*)d_in[36]; const float* dec_b2 = (const float*)d_in[37];

    // ---- workspace bump allocator (256B aligned) ----
    size_t off = 0;
    char* base = (char*)d_ws;
    auto alloc = [&](size_t bytes) -> void* {
        void* p = base + off;
        off = (off + bytes + 255) & ~(size_t)255;
        return p;
    };
    float*    nl     = (float*)   alloc((size_t)NN_ * LL * 4);
    float*    el     = (float*)   alloc((size_t)NE_ * LL * 4);
    uint16_t* nlbf   = (uint16_t*)alloc((size_t)NN_ * LL * 2);
    uint16_t* elbf   = (uint16_t*)alloc((size_t)NE_ * LL * 2);
    float*    aggr   = (float*)   alloc((size_t)NN_ * LL * 4);
    uint16_t* aggrbf = (uint16_t*)alloc((size_t)NN_ * LL * 2);
    uint16_t* Hbuf   = (uint16_t*)alloc((size_t)NE_ * LL * 2);   // reused for node hidden
    uint16_t* enW2p  = (uint16_t*)alloc((size_t)LL * LL * 2);
    uint16_t* eeW2p  = (uint16_t*)alloc((size_t)LL * LL * 2);
    uint16_t* beW1p  = (uint16_t*)alloc((size_t)SS * 3 * LL * LL * 2);
    uint16_t* beW2p  = (uint16_t*)alloc((size_t)SS * LL * LL * 2);
    uint16_t* bnW1p  = (uint16_t*)alloc((size_t)SS * 2 * LL * LL * 2);
    uint16_t* bnW2p  = (uint16_t*)alloc((size_t)SS * LL * LL * 2);

    // ---- pack weights into WMMA fragment layout ----
    auto pack = [&](const float* src, uint16_t* dst, int K) {
        int total = K * LL;
        pack_w_kernel<<<(total + 255) / 256, 256, 0, stream>>>(src, dst, K);
    };
    pack(en_W2, enW2p, LL);
    pack(ee_W2, eeW2p, LL);
    for (int s = 0; s < SS; s++) {
        pack(be_W1 + (size_t)s * 3 * LL * LL, beW1p + (size_t)s * 3 * LL * LL, 3 * LL);
        pack(be_W2 + (size_t)s * LL * LL,     beW2p + (size_t)s * LL * LL,     LL);
        pack(bn_W1 + (size_t)s * 2 * LL * LL, bnW1p + (size_t)s * 2 * LL * LL, 2 * LL);
        pack(bn_W2 + (size_t)s * LL * LL,     bnW2p + (size_t)s * LL * LL,     LL);
    }

    const int NB = NN_ / 16;   // 625
    const int EB = NE_ / 16;   // 3750

    // ---- encoders ----
    enc_node_l1_kernel<<<NN_, 128, 0, stream>>>(disp, chem, ntype, node_mean, node_std,
                                                en_W1, en_b1, Hbuf);
    gemm_relu_ln_kernel<<<NB, 256, 0, stream>>>(Hbuf, enW2p, en_b2, en_g, en_be,
                                                nullptr, nl, nlbf, nullptr, nullptr);
    enc_edge_l1_kernel<<<NE_, 128, 0, stream>>>(mesh_pos, disp, chem, send, recv,
                                                edge_mean, edge_std, ee_W1, ee_b1, Hbuf);
    gemm_relu_ln_kernel<<<EB, 256, 0, stream>>>(Hbuf, eeW2p, ee_b2, ee_g, ee_be,
                                                nullptr, el, elbf, nullptr, nullptr);

    // ---- message passing ----
    for (int s = 0; s < SS; s++) {
        zero_f32_kernel<<<(NN_ * LL + 255) / 256, 256, 0, stream>>>(aggr, NN_ * LL);
        gemm_edge_l1_kernel<<<EB, 256, 0, stream>>>(nlbf, elbf, send, recv,
                                                    beW1p + (size_t)s * 3 * LL * LL,
                                                    be_b1 + (size_t)s * LL, Hbuf);
        gemm_relu_ln_kernel<<<EB, 256, 0, stream>>>(Hbuf, beW2p + (size_t)s * LL * LL,
                                                    be_b2 + (size_t)s * LL,
                                                    be_g + (size_t)s * LL, be_be + (size_t)s * LL,
                                                    el, el, elbf, aggr, recv);
        cvt_bf16_kernel<<<(NN_ * LL + 255) / 256, 256, 0, stream>>>(aggr, aggrbf, NN_ * LL);
        gemm_node_l1_kernel<<<NB, 256, 0, stream>>>(nlbf, aggrbf,
                                                    bnW1p + (size_t)s * 2 * LL * LL,
                                                    bn_b1 + (size_t)s * LL, Hbuf);
        gemm_relu_ln_kernel<<<NB, 256, 0, stream>>>(Hbuf, bnW2p + (size_t)s * LL * LL,
                                                    bn_b2 + (size_t)s * LL,
                                                    bn_g + (size_t)s * LL, bn_be + (size_t)s * LL,
                                                    nl, nl, nlbf, nullptr, nullptr);
    }

    // ---- decoder ----
    decoder_kernel<<<(NN_ + 255) / 256, 256, 0, stream>>>(nl, dec_W1, dec_b1, dec_W2, dec_b2,
                                                          (float*)d_out);
}
